// EncoderLayer_10179072492065
// MI455X (gfx1250) — compile-verified, matrix operand order checked by reading, not run on hardware
//
#include <hip/hip_runtime.h>
#include <hip/hip_bf16.h>

typedef __attribute__((ext_vector_type(16))) __bf16 v16bf;
typedef __attribute__((ext_vector_type(8)))  __bf16 v8bf;
typedef __attribute__((ext_vector_type(4)))  __bf16 v4bf;
typedef __attribute__((ext_vector_type(8)))  float  v8f;
typedef __attribute__((__vector_size__(4 * sizeof(int)))) int i32x4;

#define LN_EPS 1e-5f
#define LDP 72   // padded LDS row length (bf16): 144 B, multiple of 16 B

#if defined(__has_builtin)
#if __has_builtin(__builtin_amdgcn_global_load_async_to_lds_b128) && \
    __has_builtin(__builtin_amdgcn_s_wait_asynccnt)
#define HAVE_ASYNC_LDS 1
#endif
#endif

#define AS1 __attribute__((address_space(1)))
#define AS3 __attribute__((address_space(3)))

__device__ __forceinline__ v8f wmma_bf16(const v16bf& a, const v16bf& b, const v8f& c) {
    return __builtin_amdgcn_wmma_f32_16x16x32_bf16(false, a, false, b, (short)0, c, false, false);
}

// Two contiguous 16-byte LDS loads -> one 16-element bf16 fragment.
__device__ __forceinline__ v16bf load_frag(const __bf16* p0, const __bf16* p1) {
    v8bf lo = *(const v8bf*)p0;
    v8bf hi = *(const v8bf*)p1;
    return __builtin_shufflevector(lo, hi, 0, 1, 2, 3, 4, 5, 6, 7,
                                   8, 9, 10, 11, 12, 13, 14, 15);
}

__device__ __forceinline__ v4bf cvt4(const float4 f) {
    v4bf b;
    b[0] = (__bf16)f.x; b[1] = (__bf16)f.y; b[2] = (__bf16)f.z; b[3] = (__bf16)f.w;
    return b;
}

// 16-byte global(bf16) -> LDS copy; async-to-LDS when toolchain supports it.
__device__ __forceinline__ void copy16B(const __bf16* g, __bf16* l) {
#ifdef HAVE_ASYNC_LDS
    __builtin_amdgcn_global_load_async_to_lds_b128(
        (AS1 i32x4*)g, (AS3 i32x4*)l, 0, 0);
#else
    *(v8bf*)l = *(const v8bf*)g;
#endif
}

__device__ __forceinline__ void async_wait() {
#ifdef HAVE_ASYNC_LDS
    __builtin_amdgcn_s_wait_asynccnt(0);
#endif
}

// ---------------------------------------------------------------------------
// fp32 -> bf16 elementwise (n multiple of 4)
// ---------------------------------------------------------------------------
__global__ __launch_bounds__(256)
void convert_bf16_kernel(const float* __restrict__ X, __bf16* __restrict__ Y, size_t n)
{
    size_t i = ((size_t)blockIdx.x * 256 + threadIdx.x) * 4;
    if (i < n) {
        float4 f = *(const float4*)&X[i];
        *(v4bf*)&Y[i] = cvt4(f);
    }
}

// ---------------------------------------------------------------------------
// W[K,N] fp32 -> Wt[N,K] bf16 (64x64 LDS-tiled transpose)
// ---------------------------------------------------------------------------
__global__ __launch_bounds__(256)
void transpose_bf16_kernel(const float* __restrict__ W, __bf16* __restrict__ Wt,
                           int K, int N)
{
    __shared__ float T[64][65];
    const int k0 = blockIdx.x * 64, n0 = blockIdx.y * 64;
    #pragma unroll
    for (int i = 0; i < 16; ++i) {
        int idx = threadIdx.x + i * 256;
        int r = idx >> 6, c = idx & 63;
        T[r][c] = W[(size_t)(k0 + r) * N + n0 + c];
    }
    __syncthreads();
    #pragma unroll
    for (int i = 0; i < 16; ++i) {
        int idx = threadIdx.x + i * 256;
        int r = idx >> 6, c = idx & 63;         // r = n-local, c = k-local
        Wt[(size_t)(n0 + r) * K + k0 + c] = (__bf16)T[c][r];
    }
}

// ---------------------------------------------------------------------------
// GEMM: C = op(A[M,K](bf16) @ Wt[N,K](bf16)^T + bias (+ resid))
// 256x64 tile / 256-thread block; 8 waves x (32x64 strip). Both LDS stages
// are straight 16-byte copies (async-to-LDS when available).
// Outputs: fp32 C and/or bf16 C (scaled) per template flags.
// ---------------------------------------------------------------------------
template<bool RELU, bool RESID, bool OUTF32, bool OUTBF16>
__global__ __launch_bounds__(256)
void gemm_bf16_wmma(const __bf16* __restrict__ A, const __bf16* __restrict__ Wt,
                    const float* __restrict__ bias, const float* __restrict__ resid,
                    float* __restrict__ Cf, __bf16* __restrict__ Cb, float oscale,
                    int M, int N, int K)
{
    __shared__ __align__(16) __bf16 As[256][LDP];  // [m][k]
    __shared__ __align__(16) __bf16 Bt[64][LDP];   // [n][k]
    const int tid  = threadIdx.x;
    const int wave = tid >> 5;
    const int lane = tid & 31;
    const int half = lane >> 4;
    const int l16  = lane & 15;
    const int kb0  = half * 8;
    const int kb1  = 16 + half * 8;
    const int m0 = blockIdx.x * 256;
    const int n0 = blockIdx.y * 64;

    v8f acc[2][4] = {};

    for (int k0 = 0; k0 < K; k0 += 64) {
        if (k0 + 64 < K) {
            __builtin_prefetch(&A[(size_t)(m0 + tid) * K + (k0 + 64)]);
            if (tid < 64)
                __builtin_prefetch(&Wt[(size_t)(n0 + tid) * K + (k0 + 64)]);
        }
        // stage A tile: 256 rows x 64 bf16 = 2048 16-byte chunks
        #pragma unroll
        for (int i = 0; i < 8; ++i) {
            int idx = tid + i * 256;
            int r = idx >> 3, c = (idx & 7) * 8;
            copy16B(&A[(size_t)(m0 + r) * K + k0 + c], &As[r][c]);
        }
        // stage B tile (already [n][k] in global): 512 chunks
        #pragma unroll
        for (int i = 0; i < 2; ++i) {
            int idx = tid + i * 256;
            int r = idx >> 3, c = (idx & 7) * 8;
            copy16B(&Wt[(size_t)(n0 + r) * K + k0 + c], &Bt[r][c]);
        }
        async_wait();
        __syncthreads();

        const int row0 = wave * 32 + l16;
        #pragma unroll
        for (int ks = 0; ks < 64; ks += 32) {
            v16bf a0 = load_frag(&As[row0][ks + kb0],      &As[row0][ks + kb1]);
            v16bf a1 = load_frag(&As[row0 + 16][ks + kb0], &As[row0 + 16][ks + kb1]);
            #pragma unroll
            for (int nt = 0; nt < 4; ++nt) {
                const int col = nt * 16 + l16;
                v16bf b = load_frag(&Bt[col][ks + kb0], &Bt[col][ks + kb1]);
                acc[0][nt] = wmma_bf16(a0, b, acc[0][nt]);
                acc[1][nt] = wmma_bf16(a1, b, acc[1][nt]);
            }
        }
        __syncthreads();
    }

    #pragma unroll
    for (int mi = 0; mi < 2; ++mi)
        #pragma unroll
        for (int nt = 0; nt < 4; ++nt) {
            const int col = n0 + nt * 16 + l16;
            const float bv = bias[col];
            #pragma unroll
            for (int r = 0; r < 8; ++r) {
                const int row = m0 + wave * 32 + mi * 16 + r + half * 8;
                float v = acc[mi][nt][r] + bv;
                if (RESID) v += resid[(size_t)row * N + col];
                if (RELU)  v = fmaxf(v, 0.0f);
                if (OUTF32)  Cf[(size_t)row * N + col] = v;
                if (OUTBF16) Cb[(size_t)row * N + col] = (__bf16)(v * oscale);
            }
        }
}

// ---------------------------------------------------------------------------
// Flash-attention per (b,h, 64-query block). dk=64, S=2048; Q pre-scaled by
// 1/sqrt(dk) at the Q-GEMM. Scores std ~1 -> plain exp is fp32-safe (softmax
// is shift-invariant); row-sums via WMMA with an all-ones B matrix.
// Q/K/V are bf16; Q/K stage as pure copies (async), V via register transpose.
// Output written bf16 (only consumed by the Wo GEMM).
// ---------------------------------------------------------------------------
__global__ __launch_bounds__(128)
void attention_wmma(const __bf16* __restrict__ Q, const __bf16* __restrict__ Km,
                    const __bf16* __restrict__ Vm, __bf16* __restrict__ O,
                    int S, int D)
{
    __shared__ __align__(16) __bf16 Qs[64][LDP];   // [q][d]
    __shared__ __align__(16) __bf16 Ks[64][LDP];   // [key][d]
    __shared__ __align__(16) __bf16 Vt[64][LDP];   // [d][key]
    __shared__ __align__(16) __bf16 Ps[64][LDP];   // probs [q][key]

    const int tid  = threadIdx.x;
    const int wave = tid >> 5;
    const int lane = tid & 31;
    const int half = lane >> 4;
    const int l16  = lane & 15;
    const int kb0  = half * 8;
    const int kb1  = 16 + half * 8;

    const int q0 = blockIdx.x * 64;
    const int bh = blockIdx.y;
    const int b  = bh >> 4;
    const int h  = bh & 15;
    const size_t rowbase = (size_t)b * S;
    const int    cbase   = h * 64;

    // stage Q block (pure bf16 copy; scale already folded in)
    #pragma unroll
    for (int i = 0; i < 4; ++i) {
        int idx = tid + i * 128;
        int r = idx >> 3, c = (idx & 7) * 8;
        copy16B(&Q[(rowbase + q0 + r) * D + cbase + c], &Qs[r][c]);
    }

    v16bf ones;
    #pragma unroll
    for (int i = 0; i < 16; ++i) ones[i] = (__bf16)1.0f;

    v8f o[4] = {};
    v8f lsum = {};

    const int rowA = wave * 16 + l16;

    for (int kb = 0; kb < S; kb += 64) {
        if (kb + 64 < S) {
            __builtin_prefetch(&Km[(rowbase + kb + 64 + tid) * D + cbase]);
            __builtin_prefetch(&Vm[(rowbase + kb + 64 + tid) * D + cbase]);
        }
        __syncthreads();
        // K row-major copy (async)
        #pragma unroll
        for (int i = 0; i < 4; ++i) {
            int idx = tid + i * 128;
            int r = idx >> 3, c = (idx & 7) * 8;
            copy16B(&Km[(rowbase + kb + r) * D + cbase + c], &Ks[r][c]);
        }
        // V transposed via 4x4 register micro-tiles
        #pragma unroll
        for (int tt = 0; tt < 2; ++tt) {
            const int t  = tid + tt * 128;
            const int tn = t & 15;      // d block
            const int tk = t >> 4;      // key block
            v4bf f[4];
            #pragma unroll
            for (int j = 0; j < 4; ++j)
                f[j] = *(const v4bf*)&Vm[(rowbase + kb + tk * 4 + j) * D + cbase + tn * 4];
            #pragma unroll
            for (int i = 0; i < 4; ++i) {
                v4bf p;
                p[0] = f[0][i]; p[1] = f[1][i]; p[2] = f[2][i]; p[3] = f[3][i];
                *(v4bf*)&Vt[tn * 4 + i][tk * 4] = p;
            }
        }
        async_wait();
        __syncthreads();

        // ---- scores = Q * K^T
        v16bf a0 = load_frag(&Qs[rowA][kb0],      &Qs[rowA][kb1]);
        v16bf a1 = load_frag(&Qs[rowA][32 + kb0], &Qs[rowA][32 + kb1]);
        v8f s[4];
        #pragma unroll
        for (int nt = 0; nt < 4; ++nt) {
            const int col = nt * 16 + l16;
            v16bf b0 = load_frag(&Ks[col][kb0],      &Ks[col][kb1]);
            v16bf b1 = load_frag(&Ks[col][32 + kb0], &Ks[col][32 + kb1]);
            v8f z = {};
            z = wmma_bf16(a0, b0, z);
            z = wmma_bf16(a1, b1, z);
            s[nt] = z;
        }

        // ---- p = exp(s), staged to LDS row-major (own strip only)
        #pragma unroll
        for (int nt = 0; nt < 4; ++nt)
            #pragma unroll
            for (int r = 0; r < 8; ++r)
                Ps[wave * 16 + r + half * 8][nt * 16 + l16] = (__bf16)__expf(s[nt][r]);

        v16bf p0 = load_frag(&Ps[rowA][kb0],      &Ps[rowA][kb1]);
        v16bf p1 = load_frag(&Ps[rowA][32 + kb0], &Ps[rowA][32 + kb1]);

        // ---- row-sums of P via WMMA with all-ones B (broadcast per column)
        lsum = wmma_bf16(p0, ones, lsum);
        lsum = wmma_bf16(p1, ones, lsum);

        // ---- O += P * V
        #pragma unroll
        for (int nt = 0; nt < 4; ++nt) {
            const int col = nt * 16 + l16;
            v16bf b0 = load_frag(&Vt[col][kb0],      &Vt[col][kb1]);
            v16bf b1 = load_frag(&Vt[col][32 + kb0], &Vt[col][32 + kb1]);
            o[nt] = wmma_bf16(p0, b0, o[nt]);
            o[nt] = wmma_bf16(p1, b1, o[nt]);
        }
    }

    // ---- normalize by row-sum, write bf16
    #pragma unroll
    for (int nt = 0; nt < 4; ++nt) {
        const int col = cbase + nt * 16 + l16;
        #pragma unroll
        for (int r = 0; r < 8; ++r) {
            const int row = q0 + wave * 16 + r + half * 8;
            O[(rowbase + row) * D + col] = (__bf16)(o[nt][r] / lsum[r]);
        }
    }
}

// ---------------------------------------------------------------------------
// Row-wise LayerNorm; optionally also emits a bf16 shadow copy for the next
// GEMM's A operand.
// ---------------------------------------------------------------------------
__global__ __launch_bounds__(256)
void layernorm_kernel(const float* __restrict__ X, const float* __restrict__ g,
                      const float* __restrict__ beta, float* __restrict__ Y,
                      __bf16* __restrict__ Yb, int D)
{
    __shared__ float s_sum[8], s_sq[8];
    const size_t row = blockIdx.x;
    const float* xr = X + row * D;

    float lsum = 0.0f, lsq = 0.0f;
    for (int i = threadIdx.x; i < D; i += 256) {
        float v = xr[i];
        lsum += v;
        lsq  += v * v;
    }
    #pragma unroll
    for (int off = 1; off < 32; off <<= 1) {
        lsum += __shfl_xor(lsum, off, 32);
        lsq  += __shfl_xor(lsq,  off, 32);
    }
    if ((threadIdx.x & 31) == 0) {
        s_sum[threadIdx.x >> 5] = lsum;
        s_sq[threadIdx.x >> 5]  = lsq;
    }
    __syncthreads();
    float tot = 0.0f, totsq = 0.0f;
    #pragma unroll
    for (int i = 0; i < 8; ++i) { tot += s_sum[i]; totsq += s_sq[i]; }
    const float mu  = tot / (float)D;
    const float inv = rsqrtf(totsq / (float)D - mu * mu + LN_EPS);

    for (int i = threadIdx.x; i < D; i += 256) {
        float v = (xr[i] - mu) * inv * g[i] + beta[i];
        Y[row * D + i] = v;
        if (Yb) Yb[row * D + i] = (__bf16)v;
    }
}

// ---------------------------------------------------------------------------
extern "C" void kernel_launch(void* const* d_in, const int* in_sizes, int n_in,
                              void* d_out, int out_size, void* d_ws, size_t ws_size,
                              hipStream_t stream)
{
    const float* x    = (const float*)d_in[0];
    const float* Wq   = (const float*)d_in[1];
    const float* bq   = (const float*)d_in[2];
    const float* Wk   = (const float*)d_in[3];
    const float* bk   = (const float*)d_in[4];
    const float* Wv   = (const float*)d_in[5];
    const float* bv   = (const float*)d_in[6];
    const float* Wo   = (const float*)d_in[7];
    const float* bo   = (const float*)d_in[8];
    const float* W1   = (const float*)d_in[9];
    const float* b1   = (const float*)d_in[10];
    const float* W2   = (const float*)d_in[11];
    const float* b2   = (const float*)d_in[12];
    const float* g1   = (const float*)d_in[13];
    const float* be1  = (const float*)d_in[14];
    const float* g2   = (const float*)d_in[15];
    const float* be2  = (const float*)d_in[16];

    const int B = 4, S = 2048, D = 1024, DFF = 4096;
    const int M = B * S;  // 8192
    const size_t MD  = (size_t)M * D;
    const size_t MDF = (size_t)M * DFF;

    char* p = (char*)d_ws;
    auto alloc = [&](size_t bytes) { char* q = p; p += (bytes + 255) & ~(size_t)255; return q; };

    __bf16* xb  = (__bf16*)alloc(MD * 2);
    __bf16* Wqt = (__bf16*)alloc((size_t)D * D * 2);
    __bf16* Wkt = (__bf16*)alloc((size_t)D * D * 2);
    __bf16* Wvt = (__bf16*)alloc((size_t)D * D * 2);
    __bf16* Wot = (__bf16*)alloc((size_t)D * D * 2);
    __bf16* W1t = (__bf16*)alloc((size_t)D * DFF * 2);
    __bf16* W2t = (__bf16*)alloc((size_t)DFF * D * 2);
    __bf16* Qb  = (__bf16*)alloc(MD * 2);
    __bf16* Kb  = (__bf16*)alloc(MD * 2);
    __bf16* Vb  = (__bf16*)alloc(MD * 2);
    __bf16* Ob  = (__bf16*)alloc(MD * 2);
    float*  h1  = (float*)alloc(MD * 4);
    float*  x2  = (float*)alloc(MD * 4);
    __bf16* x2b = (__bf16*)alloc(MD * 2);
    __bf16* F1b = (__bf16*)alloc(MDF * 2);
    float*  h2  = h1;   // h1 consumed by LN1 before FFN2 writes h2

    const dim3 gblk(256);

    // ---- one-time conversions / weight pre-transposes
    convert_bf16_kernel<<<dim3(MD / 1024), gblk, 0, stream>>>(x, xb, MD);
    transpose_bf16_kernel<<<dim3(D / 64, D / 64),   gblk, 0, stream>>>(Wq, Wqt, D, D);
    transpose_bf16_kernel<<<dim3(D / 64, D / 64),   gblk, 0, stream>>>(Wk, Wkt, D, D);
    transpose_bf16_kernel<<<dim3(D / 64, D / 64),   gblk, 0, stream>>>(Wv, Wvt, D, D);
    transpose_bf16_kernel<<<dim3(D / 64, D / 64),   gblk, 0, stream>>>(Wo, Wot, D, D);
    transpose_bf16_kernel<<<dim3(D / 64, DFF / 64), gblk, 0, stream>>>(W1, W1t, D, DFF);
    transpose_bf16_kernel<<<dim3(DFF / 64, D / 64), gblk, 0, stream>>>(W2, W2t, DFF, D);

    // ---- QKV projections (bf16 outputs; Q pre-scaled by 1/sqrt(dk))
    gemm_bf16_wmma<false, false, false, true><<<dim3(M / 256, D / 64), gblk, 0, stream>>>(
        xb, Wqt, bq, nullptr, nullptr, Qb, 0.125f, M, D, D);
    gemm_bf16_wmma<false, false, false, true><<<dim3(M / 256, D / 64), gblk, 0, stream>>>(
        xb, Wkt, bk, nullptr, nullptr, Kb, 1.0f, M, D, D);
    gemm_bf16_wmma<false, false, false, true><<<dim3(M / 256, D / 64), gblk, 0, stream>>>(
        xb, Wvt, bv, nullptr, nullptr, Vb, 1.0f, M, D, D);

    // ---- attention
    attention_wmma<<<dim3(S / 64, B * 16), dim3(128), 0, stream>>>(Qb, Kb, Vb, Ob, S, D);

    // ---- output projection + residual x  -> h1 (fp32)
    gemm_bf16_wmma<false, true, true, false><<<dim3(M / 256, D / 64), gblk, 0, stream>>>(
        Ob, Wot, bo, x, h1, nullptr, 1.0f, M, D, D);

    // ---- LN1 -> x2 (fp32 residual) + x2b (bf16 operand)
    layernorm_kernel<<<dim3(M), gblk, 0, stream>>>(h1, g1, be1, x2, x2b, D);

    // ---- FFN1 with ReLU -> bf16 only
    gemm_bf16_wmma<true, false, false, true><<<dim3(M / 256, DFF / 64), gblk, 0, stream>>>(
        x2b, W1t, b1, nullptr, nullptr, F1b, 1.0f, M, DFF, D);

    // ---- FFN2 + residual x2 -> h2 (fp32)
    gemm_bf16_wmma<false, true, true, false><<<dim3(M / 256, D / 64), gblk, 0, stream>>>(
        F1b, W2t, b2, x2, h2, nullptr, 1.0f, M, D, DFF);

    // ---- LN2 -> output
    layernorm_kernel<<<dim3(M), gblk, 0, stream>>>(h2, g2, be2, (float*)d_out, nullptr, D);
}
